// NaiveMHLA_876173328957
// MI455X (gfx1250) — matrix operand order repaired and checked by reading, not compiled, not run on hardware
//
#include <hip/hip_runtime.h>
#include <hip/hip_bf16.h>
#include <math.h>

typedef __attribute__((ext_vector_type(16))) __bf16 v16bf;
typedef __attribute__((ext_vector_type(8)))  float  v8f;

struct __align__(16) U4 { unsigned int x, y, z, w; };

union Frag {
  v16bf v;
  U4 q[2];
};

__device__ __forceinline__ unsigned short f2bf(float f) {
  union { float f; unsigned int u; } c; c.f = f;
  unsigned int r = c.u + 0x7FFFu + ((c.u >> 16) & 1u);
  return (unsigned short)(r >> 16);
}

// ---------------- CDNA5 feature detection (device pass only) ----------------
#if defined(__HIP_DEVICE_COMPILE__) && __has_builtin(__builtin_amdgcn_tensor_load_to_lds)
#define HAVE_TDM 1
#else
#define HAVE_TDM 0
#endif
#if defined(__HIP_DEVICE_COMPILE__) && __has_builtin(__builtin_amdgcn_global_load_async_to_lds_b128)
#define HAVE_ASYNC 1
#else
#define HAVE_ASYNC 0
#endif

#if HAVE_ASYNC
typedef __attribute__((ext_vector_type(4))) int ai4;
__device__ __forceinline__ void async_cp16(unsigned short* lds_dst,
                                           const unsigned short* gsrc) {
  __builtin_amdgcn_global_load_async_to_lds_b128(
      (__attribute__((address_space(1))) ai4*)(unsigned long long)gsrc,
      (__attribute__((address_space(3))) ai4*)(unsigned)(unsigned long long)lds_dst,
      0, 0);
}
__device__ __forceinline__ void wait_async0() {
#if __has_builtin(__builtin_amdgcn_s_wait_asynccnt)
  __builtin_amdgcn_s_wait_asynccnt(0);
#else
  asm volatile("s_wait_asynccnt 0x0" ::: "memory");
#endif
}
#endif

#if HAVE_TDM
typedef unsigned int u32x4 __attribute__((ext_vector_type(4)));
typedef int i32x4 __attribute__((ext_vector_type(4)));
typedef int i32x8 __attribute__((ext_vector_type(8)));
__device__ __forceinline__ void wait_tensor0() {
#if __has_builtin(__builtin_amdgcn_s_wait_tensorcnt)
  __builtin_amdgcn_s_wait_tensorcnt(0);
#else
  asm volatile("s_wait_tensorcnt 0x0" ::: "memory");
#endif
}
// 2D bf16 tile load via Tensor Data Mover, with LDS padding (pint/pamt codes)
__device__ __forceinline__ void tdm_load_2d(const void* g, const void* lds,
                                            int tile_w, int tile_h,
                                            long long stride_elems,
                                            int pint, int pamt) {
  unsigned long long ga = (unsigned long long)g;
  unsigned lo = (unsigned)(unsigned long long)lds;
  u32x4 g0; i32x8 g1;
  i32x4 g2 = {0, 0, 0, 0};
  i32x4 g3 = {0, 0, 0, 0};
  g0[0] = 1u;                                   // count=1
  g0[1] = lo;                                   // lds_addr
  g0[2] = (unsigned)ga;                         // global_addr[31:0]
  g0[3] = (unsigned)((ga >> 32) & 0x01FFFFFFull) | 0x80000000u;  // addr hi | type=2
  unsigned td0 = (unsigned)stride_elems;        // tensor_dim0
  unsigned td1 = (1u << 20);                    // tensor_dim1 (large)
  unsigned long long st = (unsigned long long)stride_elems;
  g1[0] = (int)((1u << 16) | (1u << 20) |       // data_size=2B, pad_enable
                ((unsigned)pint << 22) | ((unsigned)pamt << 25));
  g1[1] = (int)((td0 & 0xFFFFu) << 16);
  g1[2] = (int)((td0 >> 16) | ((td1 & 0xFFFFu) << 16));
  g1[3] = (int)((td1 >> 16) | ((unsigned)tile_w << 16));
  g1[4] = (int)((unsigned)tile_h);
  g1[5] = (int)(unsigned)st;
  g1[6] = (int)(unsigned)((st >> 32) & 0xFFFFull);
  g1[7] = 0;
#if __clang_major__ >= 23
  i32x8 z8 = {0, 0, 0, 0, 0, 0, 0, 0};
  __builtin_amdgcn_tensor_load_to_lds(g0, g1, g2, g3, z8, 0);
#else
  __builtin_amdgcn_tensor_load_to_lds(g0, g1, g2, g3, 0);
#endif
}
#endif

// ---------------- conversion kernels ----------------
__global__ void k_f32_to_bf16(const float* __restrict__ in,
                              unsigned short* __restrict__ out, int n) {
  int i = blockIdx.x * blockDim.x + threadIdx.x;
  if (i < n) out[i] = f2bf(in[i]);
}

// out (Cc x R) bf16 = transpose of in (R x Cc) f32
__global__ void k_transpose_f32_to_bf16(const float* __restrict__ in,
                                        unsigned short* __restrict__ out,
                                        int R, int Cc) {
  int i = blockIdx.x * blockDim.x + threadIdx.x;
  if (i < R * Cc) {
    int r = i / Cc, c = i % Cc;
    out[(size_t)c * R + r] = f2bf(in[i]);
  }
}

// ---------------- generic bf16 WMMA GEMM (Bt convention) ----------------
// C[z] = alpha * A[z] (MxK row-major, lda) @ Bt[z]^T, Bt given (NxK row-major, ldb)
#define BM 128
#define BN 128
#define BK 32
#define LDT 40  // padded LDS k-stride (elements)

template <bool OUT_BF16>
__global__ __launch_bounds__(256)
void k_gemm(const unsigned short* __restrict__ A,
            const unsigned short* __restrict__ Bt,
            void* __restrict__ Cv, int M, int N, int K,
            int lda, int ldb, int ldc,
            long long sA, long long sB, long long sC, float alpha) {
  __shared__ unsigned short As[2][BM * LDT];  // [m][k]
  __shared__ unsigned short Bs[2][BN * LDT];  // [n][k]

  const int tid  = threadIdx.x;
  const int lane = tid & 31;
  const int w    = tid >> 5;
  const int wr   = w >> 1;     // rows wr*32
  const int wc   = w & 1;      // cols wc*64
  const int half = lane >> 4;
  const int l16  = lane & 15;
  const int z    = blockIdx.z;

  const int m0 = blockIdx.y * BM;
  const int n0 = blockIdx.x * BN;

  const int tr = tid >> 1;            // tile row 0..127
  const int tk = (tid & 1) * 16;      // k chunk 0/16
  const unsigned short* Ag = A + (size_t)z * sA + (size_t)(m0 + tr) * lda + tk;
  const unsigned short* Bg = Bt + (size_t)z * sB + (size_t)(n0 + tr) * ldb + tk;

  v8f acc[2][4];
#pragma unroll
  for (int i = 0; i < 2; ++i)
#pragma unroll
    for (int j = 0; j < 4; ++j)
#pragma unroll
      for (int e = 0; e < 8; ++e) acc[i][j][e] = 0.f;

  // prologue: fill buffer 0
#if HAVE_ASYNC
  async_cp16(&As[0][tr * LDT + tk], Ag);
  async_cp16(&As[0][tr * LDT + tk + 8], Ag + 8);
  async_cp16(&Bs[0][tr * LDT + tk], Bg);
  async_cp16(&Bs[0][tr * LDT + tk + 8], Bg + 8);
  wait_async0();
#else
  {
    U4 a0 = *(const U4*)Ag, a1 = *(const U4*)(Ag + 8);
    U4 b0 = *(const U4*)Bg, b1 = *(const U4*)(Bg + 8);
    *(U4*)&As[0][tr * LDT + tk] = a0;
    *(U4*)&As[0][tr * LDT + tk + 8] = a1;
    *(U4*)&Bs[0][tr * LDT + tk] = b0;
    *(U4*)&Bs[0][tr * LDT + tk + 8] = b1;
  }
#endif
  __syncthreads();

  int buf = 0;
  for (int k0 = 0; k0 < K; k0 += BK) {
    const bool next = (k0 + BK) < K;
#if HAVE_ASYNC
    if (next) {
      async_cp16(&As[buf ^ 1][tr * LDT + tk], Ag + k0 + BK);
      async_cp16(&As[buf ^ 1][tr * LDT + tk + 8], Ag + k0 + BK + 8);
      async_cp16(&Bs[buf ^ 1][tr * LDT + tk], Bg + k0 + BK);
      async_cp16(&Bs[buf ^ 1][tr * LDT + tk + 8], Bg + k0 + BK + 8);
    }
#else
    U4 pa0, pa1, pb0, pb1;
    if (next) {
      pa0 = *(const U4*)(Ag + k0 + BK);
      pa1 = *(const U4*)(Ag + k0 + BK + 8);
      pb0 = *(const U4*)(Bg + k0 + BK);
      pb1 = *(const U4*)(Bg + k0 + BK + 8);
    }
#endif

    Frag a[2], b[4];
#pragma unroll
    for (int i = 0; i < 2; ++i) {
      int row  = wr * 32 + i * 16 + l16;
      int koff = 16 * half;
      a[i].q[0] = *(const U4*)(&As[buf][row * LDT + koff]);
      a[i].q[1] = *(const U4*)(&As[buf][row * LDT + koff + 8]);
    }
#pragma unroll
    for (int j = 0; j < 4; ++j) {
      int col = wc * 64 + j * 16 + l16;
      int kb  = 16 * half;
      b[j].q[0] = *(const U4*)(&Bs[buf][col * LDT + kb]);
      b[j].q[1] = *(const U4*)(&Bs[buf][col * LDT + kb + 8]);
    }
#pragma unroll
    for (int i = 0; i < 2; ++i)
#pragma unroll
      for (int j = 0; j < 4; ++j)
        acc[i][j] = __builtin_amdgcn_wmma_f32_16x16x32_bf16(
            false, a[i].v, false, b[j].v, (short)0, acc[i][j], false, false);

#if HAVE_ASYNC
    if (next) wait_async0();
#else
    if (next) {
      *(U4*)&As[buf ^ 1][tr * LDT + tk] = pa0;
      *(U4*)&As[buf ^ 1][tr * LDT + tk + 8] = pa1;
      *(U4*)&Bs[buf ^ 1][tr * LDT + tk] = pb0;
      *(U4*)&Bs[buf ^ 1][tr * LDT + tk + 8] = pb1;
    }
#endif
    __syncthreads();
    buf ^= 1;
  }

#pragma unroll
  for (int i = 0; i < 2; ++i) {
#pragma unroll
    for (int j = 0; j < 4; ++j) {
      int col = n0 + wc * 64 + j * 16 + l16;
#pragma unroll
      for (int r = 0; r < 8; ++r) {
        int row   = m0 + wr * 32 + i * 16 + r + 8 * half;
        float val = acc[i][j][r] * alpha;
        if (OUT_BF16) {
          unsigned short* C = (unsigned short*)Cv + (size_t)z * sC;
          C[(size_t)row * ldc + col] = f2bf(val);
        } else {
          float* C = (float*)Cv + (size_t)z * sC;
          C[(size_t)row * ldc + col] = val;
        }
      }
    }
  }
}

// ---------------- flash attention (d=512 latent, causal) ----------------
#define FD 512
#define FNH 16
#define NHFD (FNH * FD)

__global__ __launch_bounds__(256)
void k_flash(const unsigned short* __restrict__ qlat,  // (B*T, nh*512) bf16
             const unsigned short* __restrict__ ckv,   // (B*T, 512) bf16
             const unsigned short* __restrict__ ckvT,  // (B*512, T) bf16
             unsigned short* __restrict__ ctx,         // (B*T, nh*512) bf16
             int T) {
  extern __shared__ char smem[];
  unsigned short* Qs  = (unsigned short*)smem;   // [32][520]
  unsigned short* Ks  = Qs + 32 * 520;           // [64][520]  key-major
  unsigned short* KTs = Ks + 64 * 520;           // [512][72]  channel-major
  float*          Ss  = (float*)(KTs + 512 * 72);          // [32][68]
  unsigned short* Ps  = (unsigned short*)(Ss + 32 * 68);   // [32][72]
  float* rowM = (float*)(Ps + 32 * 72);
  float* rowL = rowM + 32;
  float* rowA = rowL + 32;

  const int tid  = threadIdx.x;
  const int lane = tid & 31;
  const int w    = tid >> 5;
  const int half = lane >> 4;
  const int l16  = lane & 15;

  const int q0 = blockIdx.x * 32;
  const int bh = blockIdx.y;
  const int b  = bh >> 4;
  const int h  = bh & 15;

  const size_t qbase  = ((size_t)(b * T + q0)) * NHFD + (size_t)h * FD;
  const size_t kvbase = ((size_t)(b * T)) * FD;
  const size_t ktbase = ((size_t)(b * FD)) * T;  // ckvT batch offset

  // ---- Q tile 32x512 ----
#if HAVE_TDM
  if (w == 0) tdm_load_2d(qlat + qbase, Qs, 512, 32, NHFD, 7, 3);
#elif HAVE_ASYNC
  {
    int r = tid >> 3;
    int c = (tid & 7) * 64;
    const unsigned short* src = qlat + qbase + (size_t)r * NHFD + c;
#pragma unroll
    for (int j = 0; j < 8; ++j) async_cp16(Qs + r * 520 + c + j * 8, src + j * 8);
  }
#else
  {
    int r = tid >> 3;
    int c = (tid & 7) * 64;
    const unsigned short* src = qlat + qbase + (size_t)r * NHFD + c;
#pragma unroll
    for (int j = 0; j < 8; ++j)
      *(U4*)(Qs + r * 520 + c + j * 8) = *(const U4*)(src + j * 8);
  }
#endif
  if (tid < 32) { rowM[tid] = -3.0e38f; rowL[tid] = 0.f; }

  v8f acc[2][4];
#pragma unroll
  for (int i = 0; i < 2; ++i)
#pragma unroll
    for (int j = 0; j < 4; ++j)
#pragma unroll
      for (int e = 0; e < 8; ++e) acc[i][j][e] = 0.f;

  const int nkt = (q0 + 31) / 64 + 1;
  for (int kt = 0; kt < nkt; ++kt) {
    const int s0 = kt * 64;
    __syncthreads();
    // ---- KV tiles: Ks (64x512 key-major), KTs (512x64 channel-major) ----
#if HAVE_TDM
    if (w == 0) {
      tdm_load_2d(ckv + kvbase + (size_t)s0 * FD, Ks, 512, 64, FD, 7, 3);
      tdm_load_2d(ckvT + ktbase + s0, KTs, 64, 512, T, 4, 3);
      wait_tensor0();
    }
#elif HAVE_ASYNC
    {
      int r = tid >> 2;
      int c = (tid & 3) * 128;
      const unsigned short* src = ckv + kvbase + (size_t)(s0 + r) * FD + c;
#pragma unroll
      for (int j = 0; j < 16; ++j) async_cp16(Ks + r * 520 + c + j * 8, src + j * 8);
#pragma unroll
      for (int j = 0; j < 16; ++j) {
        int idx = tid * 16 + j;
        int row = idx >> 3, ch = idx & 7;
        async_cp16(KTs + row * 72 + ch * 8,
                   ckvT + ktbase + (size_t)row * T + s0 + ch * 8);
      }
    }
    wait_async0();
#else
    {
      int r = tid >> 2;
      int c = (tid & 3) * 128;
      const unsigned short* src = ckv + kvbase + (size_t)(s0 + r) * FD + c;
#pragma unroll
      for (int j = 0; j < 16; ++j)
        *(U4*)(Ks + r * 520 + c + j * 8) = *(const U4*)(src + j * 8);
#pragma unroll
      for (int j = 0; j < 16; ++j) {
        int idx = tid * 16 + j;
        int row = idx >> 3, ch = idx & 7;
        *(U4*)(KTs + row * 72 + ch * 8) =
            *(const U4*)(ckvT + ktbase + (size_t)row * T + s0 + ch * 8);
      }
    }
#endif
    __syncthreads();

    {  // S = Q @ K^T : 8 (16x16) tiles, one per wave
      const int rs = w >> 2;
      const int ct = w & 3;
      v8f s;
#pragma unroll
      for (int e = 0; e < 8; ++e) s[e] = 0.f;
      for (int c0 = 0; c0 < FD; c0 += 32) {
        Frag a, bf;
        int row  = rs * 16 + l16;
        int koff = c0 + 16 * half;
        a.q[0] = *(const U4*)(Qs + row * 520 + koff);
        a.q[1] = *(const U4*)(Qs + row * 520 + koff + 8);
        int key = ct * 16 + l16;
        int kb  = c0 + 16 * half;
        bf.q[0] = *(const U4*)(Ks + key * 520 + kb);
        bf.q[1] = *(const U4*)(Ks + key * 520 + kb + 8);
        s = __builtin_amdgcn_wmma_f32_16x16x32_bf16(false, a.v, false, bf.v,
                                                    (short)0, s, false, false);
      }
#pragma unroll
      for (int r = 0; r < 8; ++r)
        Ss[(rs * 16 + r + 8 * half) * 68 + ct * 16 + l16] = s[r];
    }
    __syncthreads();

    {  // online softmax: 8 lanes per row, 8 cols per lane
      int r  = tid >> 3;
      int cb = tid & 7;
      float vv[8];
      float tmax = -3.0e38f;
      int qi = q0 + r;
#pragma unroll
      for (int j = 0; j < 8; ++j) {
        int c = cb + 8 * j;
        float sv = Ss[r * 68 + c];
        if (s0 + c > qi) sv = -3.0e38f;
        vv[j] = sv;
        tmax = fmaxf(tmax, sv);
      }
      tmax = fmaxf(tmax, __shfl_xor(tmax, 1, 8));
      tmax = fmaxf(tmax, __shfl_xor(tmax, 2, 8));
      tmax = fmaxf(tmax, __shfl_xor(tmax, 4, 8));
      float mold = rowM[r];
      float mnew = fmaxf(mold, tmax);
      float psum = 0.f;
#pragma unroll
      for (int j = 0; j < 8; ++j) {
        float p = __expf(vv[j] - mnew);
        psum += p;
        Ps[r * 72 + cb + 8 * j] = f2bf(p);
      }
      psum += __shfl_xor(psum, 1, 8);
      psum += __shfl_xor(psum, 2, 8);
      psum += __shfl_xor(psum, 4, 8);
      if (cb == 0) {
        float al = __expf(mold - mnew);
        rowA[r] = al;
        rowL[r] = rowL[r] * al + psum;
        rowM[r] = mnew;
      }
    }
    __syncthreads();

    {  // rescale + acc += P @ V (64-channel slice per wave)
#pragma unroll
      for (int i = 0; i < 2; ++i)
#pragma unroll
        for (int r = 0; r < 8; ++r) {
          float al = rowA[i * 16 + r + 8 * half];
#pragma unroll
          for (int j = 0; j < 4; ++j) acc[i][j][r] *= al;
        }
#pragma unroll
      for (int ks = 0; ks < 2; ++ks) {
        Frag a[2], bf[4];
#pragma unroll
        for (int i = 0; i < 2; ++i) {
          int row  = i * 16 + l16;
          int koff = ks * 32 + 16 * half;
          a[i].q[0] = *(const U4*)(Ps + row * 72 + koff);
          a[i].q[1] = *(const U4*)(Ps + row * 72 + koff + 8);
        }
#pragma unroll
        for (int j = 0; j < 4; ++j) {
          int col = w * 64 + j * 16 + l16;
          int kb  = ks * 32 + 16 * half;
          bf[j].q[0] = *(const U4*)(KTs + col * 72 + kb);
          bf[j].q[1] = *(const U4*)(KTs + col * 72 + kb + 8);
        }
#pragma unroll
        for (int i = 0; i < 2; ++i)
#pragma unroll
          for (int j = 0; j < 4; ++j)
            acc[i][j] = __builtin_amdgcn_wmma_f32_16x16x32_bf16(
                false, a[i].v, false, bf[j].v, (short)0, acc[i][j], false, false);
      }
    }
  }
  __syncthreads();

#pragma unroll
  for (int i = 0; i < 2; ++i)
#pragma unroll
    for (int r = 0; r < 8; ++r) {
      int row   = i * 16 + r + 8 * half;
      float inv = 1.0f / rowL[row];
      int t = q0 + row;
#pragma unroll
      for (int j = 0; j < 4; ++j) {
        int col = w * 64 + j * 16 + l16;
        ctx[((size_t)(b * T + t)) * NHFD + (size_t)h * FD + col] =
            f2bf(acc[i][j][r] * inv);
      }
    }
}

// ---------------- host orchestration ----------------
extern "C" void kernel_launch(void* const* d_in, const int* in_sizes, int n_in,
                              void* d_out, int out_size, void* d_ws,
                              size_t ws_size, hipStream_t stream) {
  (void)in_sizes; (void)n_in; (void)out_size; (void)ws_size;
  const int Bb = 2, T = 2048, C = 2048, Lq = 512, Lkv = 512, NH = 16, HS = 128;
  const int BT = Bb * T;

  const float* x    = (const float*)d_in[0];
  const float* Wdq  = (const float*)d_in[1];
  const float* Wuq  = (const float*)d_in[2];
  const float* Wdkv = (const float*)d_in[3];
  const float* Wuk  = (const float*)d_in[4];
  const float* Wuv  = (const float*)d_in[5];
  const float* Wo   = (const float*)d_in[6];

  float* y_out   = (float*)d_out;            // B*T*C
  float* ckv_out = y_out + (size_t)BT * C;   // B*T*Lkv

  char* p = (char*)d_ws;
  auto alloc = [&](size_t bytes) {
    char* r = p; p += (bytes + 255) & ~(size_t)255; return r;
  };
  unsigned short* xb     = (unsigned short*)alloc((size_t)BT * C * 2);
  unsigned short* Wdq_b  = (unsigned short*)alloc((size_t)Lq * C * 2);
  unsigned short* Wuq_b  = (unsigned short*)alloc((size_t)C * Lq * 2);
  unsigned short* Wdkv_b = (unsigned short*)alloc((size_t)Lkv * C * 2);
  unsigned short* Wo_b   = (unsigned short*)alloc((size_t)C * C * 2);
  unsigned short* WuqT   = (unsigned short*)alloc((size_t)Lq * C * 2);
  unsigned short* WukT   = (unsigned short*)alloc((size_t)Lkv * C * 2);
  unsigned short* WdqT   = (unsigned short*)alloc((size_t)C * Lq * 2);
  unsigned short* WuvT   = (unsigned short*)alloc((size_t)Lkv * C * 2);
  unsigned short* M1T    = (unsigned short*)alloc((size_t)Lkv * Lq * 2);
  unsigned short* KEt    = (unsigned short*)alloc((size_t)Lkv * C * 2);
  unsigned short* VEt    = (unsigned short*)alloc((size_t)C * Lkv * 2);
  unsigned short* ckvb   = (unsigned short*)alloc((size_t)BT * Lkv * 2);
  unsigned short* ckvT   = (unsigned short*)alloc((size_t)BT * Lkv * 2);
  unsigned short* t1b    = (unsigned short*)alloc((size_t)BT * Lq * 2);
  unsigned short* qb     = (unsigned short*)alloc((size_t)BT * C * 2);
  unsigned short* qlatb  = (unsigned short*)alloc((size_t)BT * NH * Lkv * 2);
  unsigned short* ctxb   = (unsigned short*)alloc((size_t)BT * NH * Lkv * 2);

  const int CT = 256;
  auto cdiv = [](int a, int b) { return (a + b - 1) / b; };
  dim3 blk(256);

  // conversions (bf16, row-major copies of torch-layout weights)
  k_f32_to_bf16<<<cdiv(BT * C, CT), CT, 0, stream>>>(x, xb, BT * C);
  k_f32_to_bf16<<<cdiv(Lq * C, CT), CT, 0, stream>>>(Wdq, Wdq_b, Lq * C);
  k_f32_to_bf16<<<cdiv(C * Lq, CT), CT, 0, stream>>>(Wuq, Wuq_b, C * Lq);
  k_f32_to_bf16<<<cdiv(Lkv * C, CT), CT, 0, stream>>>(Wdkv, Wdkv_b, Lkv * C);
  k_f32_to_bf16<<<cdiv(C * C, CT), CT, 0, stream>>>(Wo, Wo_b, C * C);
  // transposed copies
  k_transpose_f32_to_bf16<<<cdiv(C * Lq, CT), CT, 0, stream>>>(Wuq, WuqT, C, Lq);
  k_transpose_f32_to_bf16<<<cdiv(C * Lkv, CT), CT, 0, stream>>>(Wuk, WukT, C, Lkv);
  k_transpose_f32_to_bf16<<<cdiv(Lq * C, CT), CT, 0, stream>>>(Wdq, WdqT, Lq, C);
  k_transpose_f32_to_bf16<<<cdiv(C * Lkv, CT), CT, 0, stream>>>(Wuv, WuvT, C, Lkv);

  // M1^T (Lkv x Lq) = Wuk^T @ Wuq : A=WukT, Bt=WuqT
  k_gemm<true><<<dim3(512 / BN, 512 / BM, 1), blk, 0, stream>>>(
      WukT, WuqT, M1T, Lkv, Lq, C, C, C, Lq, 0, 0, 0, 1.0f);
  // KE^T (Lkv x C) = M1^T @ Wdq : A=M1T, Bt=WdqT
  k_gemm<true><<<dim3(2048 / BN, 512 / BM, 1), blk, 0, stream>>>(
      M1T, WdqT, KEt, Lkv, C, Lq, Lq, Lq, C, 0, 0, 0, 1.0f);
  // VE^T (C x Lkv) = Wo @ Wuv : A=Wo_b, Bt=WuvT
  k_gemm<true><<<dim3(512 / BN, 2048 / BM, 1), blk, 0, stream>>>(
      Wo_b, WuvT, VEt, C, Lkv, C, C, C, Lkv, 0, 0, 0, 1.0f);
  // c_kv (fp32 output) = x @ Wdkv^T : Bt = Wdkv
  k_gemm<false><<<dim3(512 / BN, 4096 / BM, 1), blk, 0, stream>>>(
      xb, Wdkv_b, ckv_out, BT, Lkv, C, C, C, Lkv, 0, 0, 0, 1.0f);
  k_f32_to_bf16<<<cdiv(BT * Lkv, CT), CT, 0, stream>>>(ckv_out, ckvb, BT * Lkv);
  // channel-major c_kv per batch: (FD x T)
  k_transpose_f32_to_bf16<<<cdiv(T * Lkv, CT), CT, 0, stream>>>(
      ckv_out, ckvT, T, Lkv);
  k_transpose_f32_to_bf16<<<cdiv(T * Lkv, CT), CT, 0, stream>>>(
      ckv_out + (size_t)T * Lkv, ckvT + (size_t)Lkv * T, T, Lkv);
  // t1 = x @ Wdq^T : Bt = Wdq
  k_gemm<true><<<dim3(512 / BN, 4096 / BM, 1), blk, 0, stream>>>(
      xb, Wdq_b, t1b, BT, Lq, C, C, C, Lq, 0, 0, 0, 1.0f);
  // q = t1 @ Wuq^T : Bt = Wuq
  k_gemm<true><<<dim3(2048 / BN, 4096 / BM, 1), blk, 0, stream>>>(
      t1b, Wuq_b, qb, BT, C, Lq, Lq, Lq, C, 0, 0, 0, 1.0f);
  // q_lat[h] = q[:,h*hs:] @ KE[h]  (Bt = KEt columns h*hs..) * 1/sqrt(hs)
  const float scale = 0.088388347648318447f;
  k_gemm<true><<<dim3(512 / BN, 4096 / BM, NH), blk, 0, stream>>>(
      qb, KEt, qlatb, BT, Lkv, HS, C, C, NH * Lkv,
      HS, HS, Lkv, scale);
  // flash attention over latent KV
  size_t shmem = (size_t)(32 * 520 + 64 * 520 + 512 * 72) * 2 + 32 * 68 * 4 +
                 32 * 72 * 2 + 3 * 32 * 4;
  (void)hipFuncSetAttribute(reinterpret_cast<const void*>(k_flash),
                            hipFuncAttributeMaxDynamicSharedMemorySize,
                            (int)shmem);
  k_flash<<<dim3(T / 32, Bb * NH), blk, shmem, stream>>>(qlatb, ckvb, ckvT,
                                                         ctxb, T);
  // y[:,h*hs:] = ctx[h] @ v_eff[h]  (Bt = VEt rows h*hs..)
  k_gemm<false><<<dim3(128 / BN, 4096 / BM, NH), blk, 0, stream>>>(
      ctxb, VEt, y_out, BT, HS, Lkv, NH * Lkv, Lkv, C,
      Lkv, (long long)HS * Lkv, HS, 1.0f);
}
